// LlamaAttention_KVmix_14843406975754
// MI455X (gfx1250) — matrix operand order repaired and checked by reading, not compiled, not run on hardware
//
#include <hip/hip_runtime.h>
#include <hip/hip_bf16.h>

typedef __attribute__((ext_vector_type(16))) _Float16 v16h;
typedef __attribute__((ext_vector_type(8)))  float    v8f;

// ---- problem constants ----
#define BB   4
#define QQ   1
#define HID  4096
#define NH   32
#define NKV  8
#define HD   128
#define SP   4096
#define SS   (SP + QQ)          // 4097
#define LK   3680               // ((S - int(0.1*S))//32)*32
#define LV   3680
#define GQ   (NH / NKV)         // 4
#define CH   256                // seq chunk per workgroup
#define NCH  17                 // ceil(4097/256)
#define ST   64                 // LDS-staged subtile (positions)
#define NST  (CH / ST)          // 4
#define KLVL 15.0f              // 2^4 - 1
#define VLVL 3.0f               // 2^2 - 1
#define SMSCALE 0.08838834764831843f  // 1/sqrt(128)

// partial record per (b*NKV+kv, chunk): 4 max + 4 sum + 4*128 acc = 520 floats
#define PART_STRIDE 520

// =====================================================================
// Kernel 1: QKV projection, C[m<4, n] = hidden[m,:] . W[n,:]
// One wave per 16x16 output tile; K-loop of 128 WMMA ops.
// Output channel layout per batch row: q[0..4095] | k[4096..5119] | v[5120..6143]
// =====================================================================
__global__ __launch_bounds__(32) void qkv_proj_wmma(
    const float* __restrict__ hidden,
    const float* __restrict__ Wq, const float* __restrict__ Wk,
    const float* __restrict__ Wv,
    float* __restrict__ qkv)     // [4][6144]
{
  const int tile = blockIdx.x;            // 0..383
  const int lane = threadIdx.x;           // 0..31
  const int mrow = lane & 15;
  const int half = lane >> 4;
  const int ncol = tile * 16 + (lane & 15);

  const float* wrow;
  if (ncol < 4096)      wrow = Wq + (size_t)ncol * HID;
  else if (ncol < 5120) wrow = Wk + (size_t)(ncol - 4096) * HID;
  else                  wrow = Wv + (size_t)(ncol - 5120) * HID;

  const float* arow = hidden + (size_t)(mrow < BB ? mrow : 0) * HID;

  v8f acc = {};
  for (int kk = 0; kk < HID; kk += 32) {
    v16h a, b;
    // A fragment: 16-bit A 16x32 layout (lane&15 = row M, half selects K phase)
#pragma unroll
    for (int g = 0; g < 2; ++g) {
      const float4 f0 = *reinterpret_cast<const float4*>(arow + kk + half * 16 + g * 8);
      const float4 f1 = *reinterpret_cast<const float4*>(arow + kk + half * 16 + g * 8 + 4);
      const int e = g * 8;
      a[e + 0] = (_Float16)f0.x; a[e + 1] = (_Float16)f0.y;
      a[e + 2] = (_Float16)f0.z; a[e + 3] = (_Float16)f0.w;
      a[e + 4] = (_Float16)f1.x; a[e + 5] = (_Float16)f1.y;
      a[e + 6] = (_Float16)f1.z; a[e + 7] = (_Float16)f1.w;
    }
    // B fragment: lanes 0-15 hold K=kk..kk+15, lanes 16-31 hold K=kk+16..kk+31
    const float* bp = wrow + kk + half * 16;
    const float4 b0 = reinterpret_cast<const float4*>(bp)[0];
    const float4 b1 = reinterpret_cast<const float4*>(bp)[1];
    const float4 b2 = reinterpret_cast<const float4*>(bp)[2];
    const float4 b3 = reinterpret_cast<const float4*>(bp)[3];
    b[0]  = (_Float16)b0.x; b[1]  = (_Float16)b0.y; b[2]  = (_Float16)b0.z; b[3]  = (_Float16)b0.w;
    b[4]  = (_Float16)b1.x; b[5]  = (_Float16)b1.y; b[6]  = (_Float16)b1.z; b[7]  = (_Float16)b1.w;
    b[8]  = (_Float16)b2.x; b[9]  = (_Float16)b2.y; b[10] = (_Float16)b2.z; b[11] = (_Float16)b2.w;
    b[12] = (_Float16)b3.x; b[13] = (_Float16)b3.y; b[14] = (_Float16)b3.z; b[15] = (_Float16)b3.w;

    acc = __builtin_amdgcn_wmma_f32_16x16x32_f16(
        /*neg_a=*/false, a, /*neg_b=*/false, b,
        /*c_mod=*/(short)0, acc, /*reuse_a=*/false, /*reuse_b=*/false);
  }

  // D layout: VGPR r -> M = r + half*8, N = lane&15
#pragma unroll
  for (int r = 0; r < 8; ++r) {
    const int mm = r + half * 8;
    if (mm < BB)
      qkv[(size_t)mm * 6144 + tile * 16 + (lane & 15)] = acc[r];
  }
}

// =====================================================================
// Kernel 2: RoPE on q and k, passthrough copy of v.
// =====================================================================
__device__ __forceinline__ float rope_mix(float x, float partner, int d, float pos) {
  const int i = d & 63;
  const float inv = __expf(-((float)(2 * i) / (float)HD) * 9.2103403719761836f); // ln(10000)
  const float f = pos * inv;
  const float c = __cosf(f), s = __sinf(f);
  const float rot = (d < 64) ? -partner : partner;
  return x * c + rot * s;
}

__global__ void rope_kernel(const float* __restrict__ qkv,
                            const int* __restrict__ pos_ids,
                            float* __restrict__ q_rot,
                            float* __restrict__ k_rot,
                            float* __restrict__ v_new)
{
  const int idx = blockIdx.x * blockDim.x + threadIdx.x;
  if (idx >= BB * (NH + 2 * NKV) * HD) return;   // 24576

  if (idx < BB * NH * HD) {                      // q: 16384
    const int b = idx / (NH * HD);
    const int rem = idx % (NH * HD);
    const int d = rem % HD;
    const float pos = (float)pos_ids[b * QQ];
    const float x = qkv[(size_t)b * 6144 + rem];
    const int prem = (d < 64) ? rem + 64 : rem - 64;
    const float p = qkv[(size_t)b * 6144 + prem];
    q_rot[idx] = rope_mix(x, p, d, pos);
  } else if (idx < BB * (NH + NKV) * HD) {       // k: 4096
    const int j = idx - BB * NH * HD;
    const int b = j / (NKV * HD);
    const int rem = j % (NKV * HD);
    const int d = rem % HD;
    const float pos = (float)pos_ids[b * QQ];
    const float x = qkv[(size_t)b * 6144 + 4096 + rem];
    const int prem = (d < 64) ? rem + 64 : rem - 64;
    const float p = qkv[(size_t)b * 6144 + 4096 + prem];
    k_rot[j] = rope_mix(x, p, d, pos);
  } else {                                       // v: 4096
    const int j = idx - BB * (NH + NKV) * HD;
    const int b = j / (NKV * HD);
    const int rem = j % (NKV * HD);
    v_new[j] = qkv[(size_t)b * 6144 + 5120 + rem];
  }
}

// =====================================================================
// Kernel 3: attention partials over one 256-position chunk.
// KV data is streamed ONCE: each 64-position subtile is staged into LDS
// with coalesced float4 loads, then min/max + in-place dequant + math all
// run out of LDS (rows padded to HD+1 -> bank-conflict-free both ways).
// grid = (NCH, B*NKV), block = 256 threads.
// =====================================================================
__global__ __launch_bounds__(256) void attn_partial(
    const float* __restrict__ pastK, const float* __restrict__ pastV,
    const float* __restrict__ q_rot, const float* __restrict__ k_rot,
    const float* __restrict__ v_new, const float* __restrict__ mask,
    float* __restrict__ part)
{
  __shared__ float tile[ST][HD + 1];             // staged K or V subtile (33 KB)
  __shared__ float q_lds[GQ][HD];
  __shared__ float gmn[ST / 32][HD], gsc[ST / 32][HD];   // K group min/scale
  __shared__ float vm[ST][HD / 32], vs[ST][HD / 32];     // V group min/scale
  __shared__ float scb[GQ][CH];
  __shared__ float hmax[GQ], hsum[GQ];

  const int chunk = blockIdx.x;
  const int bk    = blockIdx.y;          // b*NKV + kv
  const int b     = bk / NKV;
  const int kv    = bk % NKV;
  const int start = chunk * CH;
  const int valid = min(CH, SS - start);
  const int t     = threadIdx.x;

  // load grouped q heads
  for (int i = t; i < GQ * HD; i += 256) {
    const int hi = i / HD, d = i % HD;
    q_lds[hi][d] = q_rot[((size_t)b * NH + kv * GQ + hi) * HD + d];
  }

  const float* kbase = pastK + ((size_t)bk * SP) * HD;
  const float* vbase = pastV + ((size_t)bk * SP) * HD;

  // ================= pass 1: scores (K streamed once) =================
  for (int st = 0; st < NST; ++st) {
    const int sbase = start + st * ST;
    __syncthreads();                                   // tile reuse barrier
    // stage K subtile: 64*128 floats = 2048 float4, 8 per thread, coalesced
#pragma unroll
    for (int j = 0; j < 8; ++j) {
      const int i  = t + j * 256;
      const int sl = i >> 5;
      const int d0 = (i & 31) * 4;
      const int s  = sbase + sl;
      if (s < SP) {
        const float4 f =
            reinterpret_cast<const float4*>(kbase + (size_t)s * HD)[d0 >> 2];
        tile[sl][d0] = f.x; tile[sl][d0 + 1] = f.y;
        tile[sl][d0 + 2] = f.z; tile[sl][d0 + 3] = f.w;
      } else {
        tile[sl][d0] = 0.f; tile[sl][d0 + 1] = 0.f;
        tile[sl][d0 + 2] = 0.f; tile[sl][d0 + 3] = 0.f;
      }
    }
    if (sbase + ST < SP)
      __builtin_prefetch(kbase + (size_t)(sbase + ST) * HD, 0, 1);
    __syncthreads();
    // K group min/max from LDS: (ST/32)*HD = 256 pairs, one per thread
    {
      const int g = t >> 7;
      const int d = t & (HD - 1);
      if (sbase + g * 32 < LK) {
        float mn = 1e30f, mx = -1e30f;
        for (int j = 0; j < 32; ++j) {
          const float v = tile[g * 32 + j][d];
          mn = fminf(mn, v); mx = fmaxf(mx, v);
        }
        gmn[g][d] = mn;
        gsc[g][d] = (mx - mn) / KLVL;
      }
    }
    __syncthreads();
    // dequant in place (quantized region only)
#pragma unroll
    for (int j = 0; j < 8; ++j) {
      const int i  = t + j * 256;
      const int sl = i >> 5;
      const int d0 = (i & 31) * 4;
      const int s  = sbase + sl;
      if (s < LK) {
        const int g = sl >> 5;
#pragma unroll
        for (int e = 0; e < 4; ++e) {
          const float mn = gmn[g][d0 + e], sc = gsc[g][d0 + e];
          float qv = roundf((tile[sl][d0 + e] - mn) / fmaxf(sc, 1e-8f));
          qv = fminf(fmaxf(qv, 0.f), KLVL);
          tile[sl][d0 + e] = fmaf(qv, sc, mn);
        }
      }
    }
    __syncthreads();
    // scores: thread -> (head, position)
    {
      const int hi = t >> 6;
      const int sl = t & (ST - 1);
      const int s  = sbase + sl;
      const int ci = st * ST + sl;
      if (ci < valid) {
        float a = 0.f;
        if (s == SS - 1) {
          const float* kp = k_rot + (size_t)bk * HD;
          for (int d = 0; d < HD; ++d) a = fmaf(q_lds[hi][d], kp[d], a);
        } else {
          for (int d = 0; d < HD; ++d) a = fmaf(q_lds[hi][d], tile[sl][d], a);
        }
        scb[hi][ci] = fmaf(a, SMSCALE, mask[(size_t)b * SS + s]);
      } else {
        scb[hi][ci] = -1e30f;
      }
    }
  }
  __syncthreads();

  // ================= chunk-local softmax =================
  if (t < GQ) {
    float m = -1e30f;
    for (int j = 0; j < valid; ++j) m = fmaxf(m, scb[t][j]);
    hmax[t] = m;
  }
  __syncthreads();
#pragma unroll
  for (int hi = 0; hi < GQ; ++hi)
    scb[hi][t] = (t < valid) ? __expf(scb[hi][t] - hmax[hi]) : 0.f;
  __syncthreads();
  if (t < GQ) {
    float s = 0.f;
    for (int j = 0; j < valid; ++j) s += scb[t][j];
    hsum[t] = s;
  }

  // ================= pass 2: V accumulation (V streamed once) ==========
  const int d  = t & (HD - 1);
  const int h0 = (t >> 7) * 2;
  const int h1 = h0 + 1;
  float acc0 = 0.f, acc1 = 0.f;
  for (int st = 0; st < NST; ++st) {
    const int nv = min(ST, valid - st * ST);
    if (nv <= 0) break;                    // uniform across block
    const int sbase = start + st * ST;
    __syncthreads();                       // tile reuse barrier
    // stage V subtile
#pragma unroll
    for (int j = 0; j < 8; ++j) {
      const int i  = t + j * 256;
      const int sl = i >> 5;
      const int d0 = (i & 31) * 4;
      const int s  = sbase + sl;
      if (s < SP) {
        const float4 f =
            reinterpret_cast<const float4*>(vbase + (size_t)s * HD)[d0 >> 2];
        tile[sl][d0] = f.x; tile[sl][d0 + 1] = f.y;
        tile[sl][d0 + 2] = f.z; tile[sl][d0 + 3] = f.w;
      } else {
        tile[sl][d0] = 0.f; tile[sl][d0 + 1] = 0.f;
        tile[sl][d0 + 2] = 0.f; tile[sl][d0 + 3] = 0.f;
      }
    }
    __syncthreads();
    // V group min/max per (position, d-group of 32): 64*4 = 256 pairs
    {
      const int sl = t >> 2;
      const int dg = t & 3;
      if (sbase + sl < LV) {
        float mn = 1e30f, mx = -1e30f;
        for (int j = 0; j < 32; ++j) {
          const float v = tile[sl][dg * 32 + j];
          mn = fminf(mn, v); mx = fmaxf(mx, v);
        }
        vm[sl][dg] = mn;
        vs[sl][dg] = (mx - mn) / VLVL;
      }
    }
    __syncthreads();
    // dequant in place
#pragma unroll
    for (int j = 0; j < 8; ++j) {
      const int i  = t + j * 256;
      const int sl = i >> 5;
      const int d0 = (i & 31) * 4;
      const int s  = sbase + sl;
      if (s < LV) {
        const int dg = d0 >> 5;
        const float mn = vm[sl][dg], sc = vs[sl][dg];
#pragma unroll
        for (int e = 0; e < 4; ++e) {
          float qv = roundf((tile[sl][d0 + e] - mn) / fmaxf(sc, 1e-8f));
          qv = fminf(fmaxf(qv, 0.f), VLVL);
          tile[sl][d0 + e] = fmaf(qv, sc, mn);
        }
      }
    }
    __syncthreads();
    // accumulate weighted V for two heads
    for (int sl = 0; sl < nv; ++sl) {
      const int s  = sbase + sl;
      const int ci = st * ST + sl;
      const float vv = (s == SS - 1) ? v_new[(size_t)bk * HD + d] : tile[sl][d];
      acc0 = fmaf(scb[h0][ci], vv, acc0);
      acc1 = fmaf(scb[h1][ci], vv, acc1);
    }
  }

  float* pr = part + (size_t)(bk * NCH + chunk) * PART_STRIDE;
  if (t < GQ) { pr[t] = hmax[t]; pr[GQ + t] = hsum[t]; }
  pr[2 * GQ + h0 * HD + d] = acc0;
  pr[2 * GQ + h1 * HD + d] = acc1;
}

// =====================================================================
// Kernel 4: combine chunk partials -> attn_out [B, NH, HD]
// =====================================================================
__global__ __launch_bounds__(128) void attn_reduce(
    const float* __restrict__ part, float* __restrict__ attn_out)
{
  const int bh = blockIdx.x;             // b*NH + h
  const int h = bh % NH;
  const int b = bh / NH;
  const int kv = h / GQ, hi = h % GQ;
  const int bk = b * NKV + kv;
  const int d = threadIdx.x;

  float gmax = -1e30f;
  for (int c = 0; c < NCH; ++c)
    gmax = fmaxf(gmax, part[(size_t)(bk * NCH + c) * PART_STRIDE + hi]);

  float tot = 0.f, acc = 0.f;
  for (int c = 0; c < NCH; ++c) {
    const float* pr = part + (size_t)(bk * NCH + c) * PART_STRIDE;
    const float w = __expf(pr[hi] - gmax);
    tot += w * pr[GQ + hi];
    acc += w * pr[2 * GQ + hi * HD + d];
  }
  attn_out[(size_t)bh * HD + d] = acc / tot;
}

// =====================================================================
// Kernel 5: output projection out[m, n] = attn[m,:] . Wo[n,:]  (WMMA)
// =====================================================================
__global__ __launch_bounds__(32) void out_proj_wmma(
    const float* __restrict__ attn, const float* __restrict__ Wo,
    float* __restrict__ out)
{
  const int tile = blockIdx.x;            // 0..255
  const int lane = threadIdx.x;
  const int mrow = lane & 15;
  const int half = lane >> 4;
  const int ncol = tile * 16 + (lane & 15);

  const float* wrow = Wo + (size_t)ncol * (NH * HD);
  const float* arow = attn + (size_t)(mrow < BB ? mrow : 0) * (NH * HD);

  v8f acc = {};
  for (int kk = 0; kk < NH * HD; kk += 32) {
    v16h a, b;
#pragma unroll
    for (int g = 0; g < 2; ++g) {
      const float4 f0 = *reinterpret_cast<const float4*>(arow + kk + half * 16 + g * 8);
      const float4 f1 = *reinterpret_cast<const float4*>(arow + kk + half * 16 + g * 8 + 4);
      const int e = g * 8;
      a[e + 0] = (_Float16)f0.x; a[e + 1] = (_Float16)f0.y;
      a[e + 2] = (_Float16)f0.z; a[e + 3] = (_Float16)f0.w;
      a[e + 4] = (_Float16)f1.x; a[e + 5] = (_Float16)f1.y;
      a[e + 6] = (_Float16)f1.z; a[e + 7] = (_Float16)f1.w;
    }
    const float* bp = wrow + kk + half * 16;
    const float4 b0 = reinterpret_cast<const float4*>(bp)[0];
    const float4 b1 = reinterpret_cast<const float4*>(bp)[1];
    const float4 b2 = reinterpret_cast<const float4*>(bp)[2];
    const float4 b3 = reinterpret_cast<const float4*>(bp)[3];
    b[0]  = (_Float16)b0.x; b[1]  = (_Float16)b0.y; b[2]  = (_Float16)b0.z; b[3]  = (_Float16)b0.w;
    b[4]  = (_Float16)b1.x; b[5]  = (_Float16)b1.y; b[6]  = (_Float16)b1.z; b[7]  = (_Float16)b1.w;
    b[8]  = (_Float16)b2.x; b[9]  = (_Float16)b2.y; b[10] = (_Float16)b2.z; b[11] = (_Float16)b2.w;
    b[12] = (_Float16)b3.x; b[13] = (_Float16)b3.y; b[14] = (_Float16)b3.z; b[15] = (_Float16)b3.w;

    acc = __builtin_amdgcn_wmma_f32_16x16x32_f16(
        false, a, false, b, (short)0, acc, false, false);
  }

#pragma unroll
  for (int r = 0; r < 8; ++r) {
    const int mm = r + half * 8;
    if (mm < BB)
      out[(size_t)mm * HID + tile * 16 + (lane & 15)] = acc[r];
  }
}

// =====================================================================
extern "C" void kernel_launch(void* const* d_in, const int* in_sizes, int n_in,
                              void* d_out, int out_size, void* d_ws, size_t ws_size,
                              hipStream_t stream) {
  (void)in_sizes; (void)n_in; (void)out_size; (void)ws_size;
  const float* hidden = (const float*)d_in[0];
  const float* mask   = (const float*)d_in[1];
  const float* pastK  = (const float*)d_in[2];
  const float* pastV  = (const float*)d_in[3];
  const float* Wq     = (const float*)d_in[4];
  const float* Wk     = (const float*)d_in[5];
  const float* Wv     = (const float*)d_in[6];
  const float* Wo     = (const float*)d_in[7];
  const int*   pos    = (const int*)d_in[8];
  float* out = (float*)d_out;

  float* ws       = (float*)d_ws;
  float* qkv      = ws;                 // 4*6144      = 24576
  float* q_rot    = ws + 24576;         // 16384
  float* k_rot    = ws + 40960;         // 4096
  float* v_new    = ws + 45056;         // 4096
  float* attn_out = ws + 49152;         // 16384
  float* part     = ws + 65536;         // 32*17*520   = 282880

  // 1) QKV projection (WMMA): 384 tiles of 16 output channels
  qkv_proj_wmma<<<dim3(384), dim3(32), 0, stream>>>(hidden, Wq, Wk, Wv, qkv);

  // 2) RoPE + split
  rope_kernel<<<dim3(96), dim3(256), 0, stream>>>(qkv, pos, q_rot, k_rot, v_new);

  // 3) attention partials: (17 chunks) x (32 b*kv pairs)
  attn_partial<<<dim3(NCH, BB * NKV), dim3(256), 0, stream>>>(
      pastK, pastV, q_rot, k_rot, v_new, mask, part);

  // 4) combine partials
  attn_reduce<<<dim3(BB * NH), dim3(128), 0, stream>>>(part, attn_out);

  // 5) output projection (WMMA) -> d_out
  out_proj_wmma<<<dim3(256), dim3(32), 0, stream>>>(attn_out, Wo, out);
}
